// ConformerBlock_63058709840458
// MI455X (gfx1250) — matrix operand order repaired and checked by reading, not compiled, not run on hardware
//
#include <hip/hip_runtime.h>

typedef unsigned int u32;
typedef unsigned short u16;

typedef __attribute__((ext_vector_type(16))) __bf16 v16bf;
typedef __attribute__((ext_vector_type(8)))  float  v8f;

union AF { v16bf v; u32 u[8]; };
union CF { v8f  v; float f[8]; };

__device__ __forceinline__ u16 f2bf(float x) {
  u32 u = __float_as_uint(x);
  return (u16)((u + 0x7FFFu + ((u >> 16) & 1u)) >> 16);   // RNE
}
__device__ __forceinline__ float bf2f(u16 h) { return __uint_as_float(((u32)h) << 16); }
__device__ __forceinline__ float sigmoidf_(float x) { return 1.f / (1.f + __expf(-x)); }

__device__ __forceinline__ v8f wmma_bf16(const AF& a, const AF& b, v8f c) {
  return __builtin_amdgcn_wmma_f32_16x16x32_bf16(false, a.v, false, b.v, (short)0, c, false, false);
}

// gfx1250 async copy: global -> LDS, 16B per lane, tracked by ASYNCcnt.
__device__ __forceinline__ void async_b128(u32 lds_byte_addr, const void* gaddr) {
  asm volatile("global_load_async_to_lds_b128 %0, %1, off"
               :: "v"(lds_byte_addr), "v"(gaddr) : "memory");
}
__device__ __forceinline__ void wait_async0() {
  asm volatile("s_wait_asynccnt 0x0" ::: "memory");
}

// ---------------------------------------------------------------------------
// bf16-WMMA GEMM: C[M,Nc] = epi(A[M,K](bf16) @ Wt[Nc,K](bf16,transposed))
// Block tile 128x64, BK=32, 8 waves (4x2), each wave 32x32 (4 WMMAs/chunk).
// Tiles staged via GLOBAL_LOAD_ASYNC_TO_LDS_B128 (ASYNCcnt).
// EPI: 0 bias->f32 | 1 bias+swish->bf16 | 2 bias+res->f32 | 3 bias,0.5x+res->f32
//      4 no-bias->bf16
// ---------------------------------------------------------------------------
template <int EPI>
__global__ void __launch_bounds__(256)
gemm_kernel(const u16* __restrict__ A, const u16* __restrict__ Wt,
            const float* __restrict__ bias, const float* __restrict__ res,
            void* __restrict__ Cout, int M, int K, int Nc) {
  __shared__ u16 As[128 * 40];   // row stride 40 halves = 80B (16B-aligned b128 slots)
  __shared__ u16 Bs[64 * 40];    // Bs[c][k] — rows of transposed weights

  const int tid  = threadIdx.x;
  const int lane = tid & 31;
  const int wid  = tid >> 5;
  const int wm   = wid >> 1, wn = wid & 1;
  const int hi   = lane >> 4, lo = lane & 15;
  const int rowBase = blockIdx.x * 128;
  const int colBase = blockIdx.y * 64;

  CF acc[2][2];
#pragma unroll
  for (int i = 0; i < 2; i++)
#pragma unroll
    for (int j = 0; j < 2; j++)
#pragma unroll
      for (int v = 0; v < 8; v++) acc[i][j].f[v] = 0.f;

  const int aRow = tid >> 1;           // 0..127
  const int aCol = (tid & 1) * 16;     // 0 | 16 (halves)
  const int bRow = tid >> 2;           // 0..63  (C column)
  const int bCol = (tid & 3) * 8;      // 0..24  (halves)

  const u32 ldsA = (u32)(size_t)&As[aRow * 40 + aCol];
  const u32 ldsB = (u32)(size_t)&Bs[bRow * 40 + bCol];
  const u16* gA = A  + (size_t)(rowBase + aRow) * K + aCol;
  const u16* gB = Wt + (size_t)(colBase + bRow) * K + bCol;

  for (int k0 = 0; k0 < K; k0 += 32) {
    async_b128(ldsA,      gA + k0);
    async_b128(ldsA + 16, gA + k0 + 8);
    async_b128(ldsB,      gB + k0);
    wait_async0();
    __syncthreads();

    AF af[2], bfr[2];
#pragma unroll
    for (int i = 0; i < 2; i++) {
      int r = wm * 32 + i * 16 + lo;
#pragma unroll
      for (int v = 0; v < 8; v++) {
        int kk = hi * 16 + v * 2;                            // contiguous k = 16*hi + s
        af[i].u[v] = *(const u32*)&As[r * 40 + kk];
      }
    }
#pragma unroll
    for (int j = 0; j < 2; j++) {
      int c = wn * 32 + j * 16 + lo;
#pragma unroll
      for (int v = 0; v < 8; v++) {
        int kk = hi * 16 + v * 2;                            // contiguous k = 16*hi + s, same as A
        bfr[j].u[v] = *(const u32*)&Bs[c * 40 + kk];
      }
    }
#pragma unroll
    for (int i = 0; i < 2; i++)
#pragma unroll
      for (int j = 0; j < 2; j++)
        acc[i][j].v = wmma_bf16(af[i], bfr[j], acc[i][j].v);
    __syncthreads();
  }

#pragma unroll
  for (int i = 0; i < 2; i++) {
#pragma unroll
    for (int j = 0; j < 2; j++) {
      int gc = colBase + wn * 32 + j * 16 + lo;
      float bv = (EPI == 4) ? 0.f : bias[gc];
#pragma unroll
      for (int v = 0; v < 8; v++) {
        int gr = rowBase + wm * 32 + i * 16 + v + 8 * hi;    // ISA C/D layout
        float val = acc[i][j].f[v] + bv;
        size_t idx = (size_t)gr * Nc + gc;
        if (EPI == 0)      ((float*)Cout)[idx] = val;
        else if (EPI == 1) ((u16*)Cout)[idx] = f2bf(val * sigmoidf_(val));
        else if (EPI == 2) ((float*)Cout)[idx] = res[idx] + val;
        else if (EPI == 3) ((float*)Cout)[idx] = res[idx] + 0.5f * val;
        else               ((u16*)Cout)[idx] = f2bf(val);
      }
    }
  }
}

// ---------------------------------------------------------------------------
// LayerNorm over last dim (256 or 512), one wave per token, optional swish,
// output either bf16 (feeds a WMMA GEMM) or f32 (final output).
// ---------------------------------------------------------------------------
template <int DIM, bool SWISH, bool OUTBF>
__global__ void __launch_bounds__(256)
ln_kernel(const float* __restrict__ x, const float* __restrict__ g,
          const float* __restrict__ b, void* __restrict__ out) {
  const int lane  = threadIdx.x & 31;
  const int token = blockIdx.x * 8 + (threadIdx.x >> 5);
  const int PER   = DIM / 32;
  const float* xp = x + (size_t)token * DIM + lane * PER;
  float v[PER];
#pragma unroll
  for (int q = 0; q < PER / 4; q++) {
    float4 f = ((const float4*)xp)[q];
    v[q * 4 + 0] = f.x; v[q * 4 + 1] = f.y; v[q * 4 + 2] = f.z; v[q * 4 + 3] = f.w;
  }
  float s = 0.f;
#pragma unroll
  for (int i = 0; i < PER; i++) s += v[i];
#pragma unroll
  for (int o = 16; o >= 1; o >>= 1) s += __shfl_xor(s, o, 32);
  float mean = s / (float)DIM;
  float vs = 0.f;
#pragma unroll
  for (int i = 0; i < PER; i++) { float d = v[i] - mean; vs += d * d; }
#pragma unroll
  for (int o = 16; o >= 1; o >>= 1) vs += __shfl_xor(vs, o, 32);
  float rstd = rsqrtf(vs / (float)DIM + 1e-3f);
#pragma unroll
  for (int i = 0; i < PER; i++) {
    float y = (v[i] - mean) * rstd * g[lane * PER + i] + b[lane * PER + i];
    if (SWISH) y = y * sigmoidf_(y);
    size_t oi = (size_t)token * DIM + lane * PER + i;
    if (OUTBF) ((u16*)out)[oi] = f2bf(y);
    else       ((float*)out)[oi] = y;
  }
}

// ---------------------------------------------------------------------------
// Fused attention: one block per (b,h, 16-query tile). Scores (QK^T + rel via
// Pt = Q @ rel_emb^T WMMA + diagonal shuffle gather) -> LDS bf16 -> softmax ->
// P @ V with WMMA. rel index t = clip(i-j,-512,512)+512. Output bf16.
// ---------------------------------------------------------------------------
__global__ void __launch_bounds__(256)
attn_kernel(const u16* __restrict__ qb, const u16* __restrict__ kvb,
            const u16* __restrict__ relb, u16* __restrict__ outp) {
  const int N = 1024, IA = 512, DH = 64;
  __shared__ u16   SP[16 * 1024];       // scores, then P (bf16)
  __shared__ float Opart[2][16][64];
  __shared__ float red[16][16];
  __shared__ float rowm[16];
  __shared__ float rowl[16];

  const int b  = blockIdx.x >> 3, h = blockIdx.x & 7;
  const int i0 = blockIdx.y * 16;
  const int tid = threadIdx.x, lane = tid & 31, w = tid >> 5;
  const int hi = lane >> 4, lo = lane & 15;
  const float scale = 0.125f;           // DH^-0.5

  AF aq[2];
  {
    const u16* qrow = qb + ((size_t)(b * N + i0 + lo)) * IA + h * DH;
#pragma unroll
    for (int kc = 0; kc < 2; kc++)
#pragma unroll
      for (int v = 0; v < 8; v++) {
        int kk = kc * 32 + hi * 16 + v * 2;
        aq[kc].u[v] = *(const u32*)&qrow[kk];
      }
  }

  // Phase 1: this wave computes 128 keys of the score strip
  for (int kt = 0; kt < 8; kt++) {
    int j0 = w * 128 + kt * 16;
    CF d, r0, r1;
#pragma unroll
    for (int v = 0; v < 8; v++) { d.f[v] = 0.f; r0.f[v] = 0.f; r1.f[v] = 0.f; }
    const u16* krow = kvb + ((size_t)(b * N + j0 + lo)) * (2 * IA) + h * DH;  // K part
#pragma unroll
    for (int kc = 0; kc < 2; kc++) {
      AF bk;
#pragma unroll
      for (int v = 0; v < 8; v++) {
        int kk = kc * 32 + hi * 16 + v * 2;
        bk.u[v] = *(const u32*)&krow[kk];
      }
      d.v = wmma_bf16(aq[kc], bk, d.v);
#pragma unroll
      for (int mt = 0; mt < 2; mt++) {
        int m  = mt * 16 + lo;
        int tix = i0 - j0 + m - 15;
        tix = (tix < -512) ? -512 : (tix > 512 ? 512 : tix);
        tix += 512;
        const u16* erow = relb + (size_t)tix * DH;
        AF be;
#pragma unroll
        for (int v = 0; v < 8; v++) {
          int kk = kc * 32 + hi * 16 + v * 2;
          be.u[v] = *(const u32*)&erow[kk];
        }
        if (mt == 0) r0.v = wmma_bf16(aq[kc], be, r0.v);
        else         r1.v = wmma_bf16(aq[kc], be, r1.v);
      }
    }
#pragma unroll
    for (int v = 0; v < 8; v++) {
      int u = v + 8 * hi;
      int c = 15 + u - lo;                 // 0..30
      int src = (c & 15) + 16 * hi;
      float rva = __shfl(r0.f[v], src, 32);
      float rvb = __shfl(r1.f[v], src, 32);
      float rv = (c < 16) ? rva : rvb;
      SP[u * 1024 + (j0 + lo)] = f2bf((d.f[v] + rv) * scale);
    }
  }
  __syncthreads();

  // Phase 2: softmax over 1024 keys (16 rows x 16 threads/row x 64 cols)
  {
    int row = tid >> 4, sub = tid & 15;
    float mx = -3.0e38f;
    for (int c = 0; c < 64; c++) mx = fmaxf(mx, bf2f(SP[row * 1024 + sub * 64 + c]));
    red[row][sub] = mx;
    __syncthreads();
    if (sub == 0) {
      float m = red[row][0];
      for (int i = 1; i < 16; i++) m = fmaxf(m, red[row][i]);
      rowm[row] = m;
    }
    __syncthreads();
    float m = rowm[row];
    float ls = 0.f;
    for (int c = 0; c < 64; c++) {
      int idx = row * 1024 + sub * 64 + c;
      float p = __expf(bf2f(SP[idx]) - m);
      SP[idx] = f2bf(p);
      ls += p;
    }
    red[row][sub] = ls;
    __syncthreads();
    if (sub == 0) {
      float t = 0.f;
      for (int i = 0; i < 16; i++) t += red[row][i];
      rowl[row] = t;
    }
    __syncthreads();
  }

  // Phase 3: O = P @ V. wave -> d-tile (w&3), key half (w>>2), 16 K-chunks.
  {
    int dt = w & 3, kh = w >> 2;
    CF o;
#pragma unroll
    for (int v = 0; v < 8; v++) o.f[v] = 0.f;
    for (int kc = 0; kc < 16; kc++) {
      int kb = kh * 512 + kc * 32;
      AF ap, bv;
#pragma unroll
      for (int v = 0; v < 8; v++) {
        int kk = kb + hi * 16 + v * 2;
        ap.u[v] = *(const u32*)&SP[lo * 1024 + kk];
      }
#pragma unroll
      for (int v = 0; v < 8; v++) {
        int kk = kb + hi * 16 + v * 2;
        const u16* v0 = kvb + ((size_t)(b * N + kk)) * (2 * IA) + IA + h * DH + dt * 16 + lo;
        const u16* v1 = v0 + 2 * IA;
        bv.u[v] = (u32)(*v0) | ((u32)(*v1) << 16);
      }
      o.v = wmma_bf16(ap, bv, o.v);
    }
#pragma unroll
    for (int v = 0; v < 8; v++) {
      int u = v + 8 * hi;
      Opart[kh][u][dt * 16 + lo] = o.f[v];
    }
  }
  __syncthreads();

#pragma unroll
  for (int r = 0; r < 4; r++) {
    int e = tid + 256 * r;
    int u = e >> 6, dd = e & 63;
    float s = (Opart[0][u][dd] + Opart[1][u][dd]) / rowl[u];
    outp[((size_t)(b * N + i0 + u)) * IA + h * DH + dd] = f2bf(s);
  }
}

// ---------------------------------------------------------------------------
// Weight transpose + fp32->bf16: w[K][Nc] -> wt[Nc][K] (done once per launch)
// ---------------------------------------------------------------------------
__global__ void __launch_bounds__(256)
wtrans_kernel(const float* __restrict__ w, u16* __restrict__ wt, int K, int Nc) {
  __shared__ float tile[32][33];
  int k0 = blockIdx.x * 32, c0 = blockIdx.y * 32;
  int tx = threadIdx.x & 31, ty = threadIdx.x >> 5;   // ty 0..7
  for (int r = ty; r < 32; r += 8)
    tile[r][tx] = w[(size_t)(k0 + r) * Nc + c0 + tx];
  __syncthreads();
  for (int r = ty; r < 32; r += 8)
    wt[(size_t)(c0 + r) * K + k0 + tx] = f2bf(tile[tx][r]);
}

// ---------------------------------------------------------------------------
// Small VALU kernels
// ---------------------------------------------------------------------------
__global__ void __launch_bounds__(256)
glu_kernel(const float* __restrict__ in, float* __restrict__ out) {
  int idx = blockIdx.x * 256 + threadIdx.x;            // < 8192*512
  int row = idx >> 9, c = idx & 511;
  float a = in[(size_t)row * 1024 + c];
  float g = in[(size_t)row * 1024 + 512 + c];
  out[idx] = a * sigmoidf_(g);
}

__global__ void __launch_bounds__(256)
dwconv_kernel(const float* __restrict__ in, const float* __restrict__ wgt,
              const float* __restrict__ bias, float* __restrict__ out) {
  int bn = blockIdx.x;                                 // b*1024 + n
  int n = bn & 1023;
#pragma unroll
  for (int half = 0; half < 2; half++) {
    int c = threadIdx.x + half * 256;
    float acc = bias[c];
    for (int k = 0; k < 31; k++) {
      int nn = n + k - 15;
      if (nn >= 0 && nn < 1024)
        acc += in[((size_t)(bn + k - 15)) * 512 + c] * wgt[k * 512 + c];
    }
    out[(size_t)bn * 512 + c] = acc;
  }
}

__global__ void __launch_bounds__(256)
cvt_bf16_kernel(const float* __restrict__ in, u16* __restrict__ out, int n) {
  int i = blockIdx.x * 256 + threadIdx.x;
  if (i < n) out[i] = f2bf(in[i]);
}

// ---------------------------------------------------------------------------
extern "C" void kernel_launch(void* const* d_in, const int* in_sizes, int n_in,
                              void* d_out, int out_size, void* d_ws, size_t ws_size,
                              hipStream_t stream) {
  (void)in_sizes; (void)n_in; (void)out_size; (void)ws_size;
  const float* x      = (const float*)d_in[0];
  const float* ff1_g  = (const float*)d_in[1];
  const float* ff1_b  = (const float*)d_in[2];
  const float* ff1_w1 = (const float*)d_in[3];
  const float* ff1_b1 = (const float*)d_in[4];
  const float* ff1_w2 = (const float*)d_in[5];
  const float* ff1_b2 = (const float*)d_in[6];
  const float* attn_g = (const float*)d_in[7];
  const float* attn_b = (const float*)d_in[8];
  const float* wq     = (const float*)d_in[9];
  const float* wkv    = (const float*)d_in[10];
  const float* wo     = (const float*)d_in[11];
  const float* bo     = (const float*)d_in[12];
  const float* rel    = (const float*)d_in[13];
  const float* conv_g = (const float*)d_in[14];
  const float* conv_b = (const float*)d_in[15];
  const float* pw1_w  = (const float*)d_in[16];
  const float* pw1_b  = (const float*)d_in[17];
  const float* dw_w   = (const float*)d_in[18];
  const float* dw_b   = (const float*)d_in[19];
  const float* ln2_g  = (const float*)d_in[20];
  const float* ln2_b  = (const float*)d_in[21];
  const float* pw2_w  = (const float*)d_in[22];
  const float* pw2_b  = (const float*)d_in[23];
  const float* ff2_g  = (const float*)d_in[24];
  const float* ff2_b  = (const float*)d_in[25];
  const float* ff2_w1 = (const float*)d_in[26];
  const float* ff2_b1 = (const float*)d_in[27];
  const float* ff2_w2 = (const float*)d_in[28];
  const float* ff2_b2 = (const float*)d_in[29];
  const float* post_g = (const float*)d_in[30];
  const float* post_b = (const float*)d_in[31];

  const int M = 8192;                       // B*N tokens
  char* ws = (char*)d_ws;
  size_t off = 0;
  auto alloc = [&](size_t bytes) -> void* {
    void* p = ws + off;
    off += (bytes + 255) & ~(size_t)255;
    return p;
  };
  float* x_cur  = (float*)alloc((size_t)M * 256 * 4);
  u16*   h_bf   = (u16*)  alloc((size_t)M * 512 * 2);   // LN outputs (bf16)
  float* mid    = (float*)alloc((size_t)M * 1024 * 4);  // f32 (pw1) or bf16 (ff mid)
  u16*   q_bf   = (u16*)  alloc((size_t)M * 512 * 2);
  u16*   kv_bf  = (u16*)  alloc((size_t)M * 1024 * 2);
  u16*   rel_bf = (u16*)  alloc((size_t)1025 * 64 * 2);
  u16*   attn_o = (u16*)  alloc((size_t)M * 512 * 2);
  float* gbuf   = (float*)alloc((size_t)M * 512 * 4);
  float* cbuf   = (float*)alloc((size_t)M * 512 * 4);
  // transposed bf16 weights [Nc][K]
  u16* ff1_w1t = (u16*)alloc((size_t)256 * 1024 * 2);
  u16* ff1_w2t = (u16*)alloc((size_t)1024 * 256 * 2);
  u16* wqt     = (u16*)alloc((size_t)256 * 512 * 2);
  u16* wkvt    = (u16*)alloc((size_t)256 * 1024 * 2);
  u16* wot     = (u16*)alloc((size_t)512 * 256 * 2);
  u16* pw1_wt  = (u16*)alloc((size_t)256 * 1024 * 2);
  u16* pw2_wt  = (u16*)alloc((size_t)512 * 256 * 2);
  u16* ff2_w1t = (u16*)alloc((size_t)256 * 1024 * 2);
  u16* ff2_w2t = (u16*)alloc((size_t)1024 * 256 * 2);

  // one-time (per launch) weight transpose/convert + rel table convert
  wtrans_kernel<<<dim3(8, 32),  256, 0, stream>>>(ff1_w1, ff1_w1t, 256, 1024);
  wtrans_kernel<<<dim3(32, 8),  256, 0, stream>>>(ff1_w2, ff1_w2t, 1024, 256);
  wtrans_kernel<<<dim3(8, 16),  256, 0, stream>>>(wq,     wqt,     256, 512);
  wtrans_kernel<<<dim3(8, 32),  256, 0, stream>>>(wkv,    wkvt,    256, 1024);
  wtrans_kernel<<<dim3(16, 8),  256, 0, stream>>>(wo,     wot,     512, 256);
  wtrans_kernel<<<dim3(8, 32),  256, 0, stream>>>(pw1_w,  pw1_wt,  256, 1024);
  wtrans_kernel<<<dim3(16, 8),  256, 0, stream>>>(pw2_w,  pw2_wt,  512, 256);
  wtrans_kernel<<<dim3(8, 32),  256, 0, stream>>>(ff2_w1, ff2_w1t, 256, 1024);
  wtrans_kernel<<<dim3(32, 8),  256, 0, stream>>>(ff2_w2, ff2_w2t, 1024, 256);
  cvt_bf16_kernel<<<257, 256, 0, stream>>>(rel, rel_bf, 1025 * 64);

  // --- FF module 1: x = x + 0.5*(swish(LN(x)W1+b1)W2+b2) ---
  ln_kernel<256, false, true><<<1024, 256, 0, stream>>>(x, ff1_g, ff1_b, h_bf);
  gemm_kernel<1><<<dim3(64, 16), 256, 0, stream>>>(h_bf, ff1_w1t, ff1_b1, nullptr, mid, M, 256, 1024);
  gemm_kernel<3><<<dim3(64, 4),  256, 0, stream>>>((u16*)mid, ff1_w2t, ff1_b2, x, x_cur, M, 1024, 256);

  // --- Attention ---
  ln_kernel<256, false, true><<<1024, 256, 0, stream>>>(x_cur, attn_g, attn_b, h_bf);
  gemm_kernel<4><<<dim3(64, 8),  256, 0, stream>>>(h_bf, wqt,  nullptr, nullptr, q_bf,  M, 256, 512);
  gemm_kernel<4><<<dim3(64, 16), 256, 0, stream>>>(h_bf, wkvt, nullptr, nullptr, kv_bf, M, 256, 1024);
  attn_kernel<<<dim3(64, 64), 256, 0, stream>>>(q_bf, kv_bf, rel_bf, attn_o);
  gemm_kernel<2><<<dim3(64, 4), 256, 0, stream>>>(attn_o, wot, bo, x_cur, x_cur, M, 512, 256);

  // --- Conv module ---
  ln_kernel<256, false, true><<<1024, 256, 0, stream>>>(x_cur, conv_g, conv_b, h_bf);
  gemm_kernel<0><<<dim3(64, 16), 256, 0, stream>>>(h_bf, pw1_wt, pw1_b, nullptr, mid, M, 256, 1024);
  glu_kernel<<<16384, 256, 0, stream>>>(mid, gbuf);
  dwconv_kernel<<<8192, 256, 0, stream>>>(gbuf, dw_w, dw_b, cbuf);
  ln_kernel<512, true, true><<<1024, 256, 0, stream>>>(cbuf, ln2_g, ln2_b, h_bf);
  gemm_kernel<2><<<dim3(64, 4), 256, 0, stream>>>(h_bf, pw2_wt, pw2_b, x_cur, x_cur, M, 512, 256);

  // --- FF module 2 ---
  ln_kernel<256, false, true><<<1024, 256, 0, stream>>>(x_cur, ff2_g, ff2_b, h_bf);
  gemm_kernel<1><<<dim3(64, 16), 256, 0, stream>>>(h_bf, ff2_w1t, ff2_b1, nullptr, mid, M, 256, 1024);
  gemm_kernel<3><<<dim3(64, 4),  256, 0, stream>>>((u16*)mid, ff2_w2t, ff2_b2, x_cur, x_cur, M, 1024, 256);

  // --- Post LN -> output (fp32) ---
  ln_kernel<256, false, false><<<1024, 256, 0, stream>>>(x_cur, post_g, post_b, (float*)d_out);
}
